// CausalMaskedAttention_47562467836374
// MI455X (gfx1250) — matrix operand order repaired and checked by reading, not compiled, not run on hardware
//
#include <hip/hip_runtime.h>
#include <cstdint>

// CausalMaskedAttention for MI455X (gfx1250, wave32, WMMA + async-to-LDS).
// B=4, N=4096, D_IN=D_OUT=768, fp32 in/out, bf16 WMMA internally.
//
// Workspace layout (bytes), total ~99.4 MB:
//   Xb  [B*N,768] bf16 @ 0
//   Qb  [B*N,768] bf16 @ 24MB
//   Kb  [B*N,768] bf16 @ 48MB
//   Vtb [B][768][N] bf16 (transposed V) @ 72MB
//   Wqb/Wkb/Wvb [768,768] bf16 @ 96MB

typedef __attribute__((ext_vector_type(16))) __bf16 v16bf;
typedef __attribute__((ext_vector_type(8)))  float  v8f;

union V16 { v16bf v; uint4 u[2]; };

#define NB   4
#define NSEQ 4096
#define ND   768

// per-buffer LDS: K chunk 32x768 bf16 (48KB) + Vt chunk 768x32 bf16 (48KB)
#define KBYTES   49152
#define BUFBYTES 98304
#define PS_OFF   (2 * BUFBYTES)
#define SMEM_SZ  (2 * BUFBYTES + 4096)

// ---------------------------------------------------------------- convert
__global__ __launch_bounds__(256) void cvt_f32_bf16(const float* __restrict__ src,
                                                    __bf16* __restrict__ dst, int n8) {
  int i = blockIdx.x * 256 + threadIdx.x;
  if (i >= n8) return;
  const float4* s = (const float4*)src + (size_t)i * 2;
  float4 a = s[0], b = s[1];
  union { __bf16 h[8]; uint4 u; } pk;
  pk.h[0] = (__bf16)a.x; pk.h[1] = (__bf16)a.y; pk.h[2] = (__bf16)a.z; pk.h[3] = (__bf16)a.w;
  pk.h[4] = (__bf16)b.x; pk.h[5] = (__bf16)b.y; pk.h[6] = (__bf16)b.z; pk.h[7] = (__bf16)b.w;
  ((uint4*)dst)[i] = pk.u;
}

// ---------------------------------------------------------------- QKV projection
// One 16x16 output tile per wave; which in {Q,K,V}; K-loop = 768/32 = 24 WMMAs.
__global__ __launch_bounds__(256) void qkv_proj(const __bf16* __restrict__ Xb,
    const __bf16* __restrict__ Wq, const __bf16* __restrict__ Wk, const __bf16* __restrict__ Wv,
    __bf16* __restrict__ Qb, __bf16* __restrict__ Kb, __bf16* __restrict__ Vtb)
{
  const int lane = threadIdx.x & 31;
  const int wv   = threadIdx.x >> 5;
  const int w    = blockIdx.x * 8 + wv;            // global wave-tile id
  const int which = w / 49152;                     // 1024 mtiles * 48 otiles
  const int r  = w % 49152;
  const int mt = r / 48, ot = r % 48;
  const int l15 = lane & 15, hi = lane >> 4;

  const __bf16* W = (which == 0) ? Wq : ((which == 1) ? Wk : Wv);
  const __bf16* aptr = Xb + (size_t)(mt * 16 + l15) * ND;   // A row (M = lane&15)
  const __bf16* bptr = W  + (size_t)(ot * 16 + l15) * ND;   // B col (N = lane&15), contig K

  v8f acc = {};
  #pragma unroll
  for (int dd = 0; dd < 24; ++dd) {
    V16 a, b;
    const int ad = dd * 32 + hi * 8;     // A: K = {0..7,16..23} / {8..15,24..31}
    a.u[0] = *(const uint4*)(aptr + ad);
    a.u[1] = *(const uint4*)(aptr + ad + 16);
    const int bd = dd * 32 + hi * 8;     // B: K = {0..7,16..23} / {8..15,24..31}
    b.u[0] = *(const uint4*)(bptr + bd);
    b.u[1] = *(const uint4*)(bptr + bd + 16);
    acc = __builtin_amdgcn_wmma_f32_16x16x32_bf16(false, a.v, false, b.v,
                                                  (short)0, acc, false, false);
  }

  if (which < 2) {                         // Q,K row-major bf16
    __bf16* O = (which == 0) ? Qb : Kb;
    #pragma unroll
    for (int j = 0; j < 8; ++j) {
      const int row = mt * 16 + j + hi * 8;          // D: M = j (+8 for hi half)
      O[(size_t)row * ND + ot * 16 + l15] = (__bf16)acc[j];
    }
  } else {                                 // V transposed: [b][768][N]
    const int rowbase = mt * 16 + hi * 8;
    const int b  = rowbase >> 12;          // /4096 (tiles never straddle batches)
    const int nb = rowbase & 4095;
    union { __bf16 h[8]; uint4 u; } pk;
    #pragma unroll
    for (int j = 0; j < 8; ++j) pk.h[j] = (__bf16)acc[j];
    *(uint4*)(Vtb + (size_t)b * ND * NSEQ + (size_t)(ot * 16 + l15) * NSEQ + nb) = pk.u;
  }
}

// ---------------------------------------------------------------- async prefetch
// Stage one 32-key chunk (K 32x768 + Vt 768x32, bf16) into LDS buffer `buf`
// with GLOBAL_LOAD_ASYNC_TO_LDS_B128 (ASYNCcnt-tracked DMA-style copies).
static __device__ __forceinline__ void prefetch_kv(
    const __bf16* Kb_b, const __bf16* Vtb_b, uint32_t smemBase,
    int tid, int kbase, int buf)
{
  // K chunk: 48KB contiguous in global (rows are consecutive)
  uint32_t lK = smemBase + (uint32_t)buf * BUFBYTES + (uint32_t)tid * 16;
  uint64_t gK = (uint64_t)(uintptr_t)(Kb_b + (size_t)kbase * ND) + (uint64_t)tid * 16;
  #pragma unroll
  for (int j = 0; j < 24; ++j) {
    asm volatile("global_load_async_to_lds_b128 %0, %1, off"
                 :: "v"(lK), "v"(gK) : "memory");
    lK += 2048; gK += 2048;
  }
  // Vt chunk: per step of 128 b128s, o advances by 32 -> LDS +2048B, global +256KB
  const int o0 = tid >> 2, p0 = (tid & 3) * 8;
  uint32_t lV = smemBase + (uint32_t)buf * BUFBYTES + KBYTES + (uint32_t)(o0 * 32 + p0) * 2;
  uint64_t gV = (uint64_t)(uintptr_t)(Vtb_b + (size_t)o0 * NSEQ + kbase + p0);
  #pragma unroll
  for (int j = 0; j < 24; ++j) {
    asm volatile("global_load_async_to_lds_b128 %0, %1, off"
                 :: "v"(lV), "v"(gV) : "memory");
    lV += 2048; gV += 262144;
  }
}

// ---------------------------------------------------------------- flash attention
// 4 waves/WG, wave = 16 q rows, WG = 64 q rows. Double-buffered 32-key chunks:
// async loads for chunk c+1 overlap the 96 WMMAs of chunk c.
__global__ __launch_bounds__(128) void flash_attn(
    const __bf16* __restrict__ Qb, const __bf16* __restrict__ Kb,
    const __bf16* __restrict__ Vtb, float* __restrict__ out)
{
  extern __shared__ char smem[];
  const uint32_t smemBase = (uint32_t)(uintptr_t)smem;   // LDS aperture: low 32 bits = offset
  __bf16* Ps = (__bf16*)(smem + PS_OFF);                 // [4][16][32] 4KB

  const int tid  = threadIdx.x;
  const int lane = tid & 31;
  const int wv   = tid >> 5;
  const int l15  = lane & 15;
  const int hi   = lane >> 4;
  const int hi8  = hi * 8;

  const int batch  = blockIdx.x >> 6;
  const int qblock = (blockIdx.x & 63) * 64;
  const int qb     = qblock + wv * 16;

  const size_t bofs = (size_t)batch * NSEQ * ND;    // same for QK rows, Vt, out
  const __bf16* Kb_b  = Kb + bofs;
  const __bf16* Vtb_b = Vtb + bofs;
  const __bf16* Qrow  = Qb + bofs + (size_t)(qb + l15) * ND;

  float m_run[8], l_run[8];
  #pragma unroll
  for (int j = 0; j < 8; ++j) { m_run[j] = -__builtin_inff(); l_run[j] = 0.f; }
  v8f oacc[48];
  #pragma unroll
  for (int ot = 0; ot < 48; ++ot) oacc[ot] = (v8f){};

  const float scale = 0.03608439182435161f;   // 1/sqrt(768) (d_in, per reference)
  const float LOG2E = 1.4426950408889634f;
  const int   nchunks = (qblock + 64) >> 5;

  // warm-up: stage chunk 0 into buffer 0
  prefetch_kv(Kb_b, Vtb_b, smemBase, tid, 0, 0);
  asm volatile("s_wait_asynccnt 0" ::: "memory");
  __syncthreads();

  for (int c = 0; c < nchunks; ++c) {
    const int buf   = c & 1;
    const int kbase = c * 32;
    // async prefetch of next chunk overlaps this chunk's WMMAs
    if (c + 1 < nchunks)
      prefetch_kv(Kb_b, Vtb_b, smemBase, tid, kbase + 32, buf ^ 1);

    const __bf16* Ks = (const __bf16*)(smem + buf * BUFBYTES);           // [32][768]
    const __bf16* Vs = (const __bf16*)(smem + buf * BUFBYTES + KBYTES);  // [768][32]

    if (kbase <= qb + 15) {                 // chunk intersects causal region
      // ---- S = Q K^T : two 16x16 tiles, shared A (Q) fragment
      v8f s0 = {}, s1 = {};
      for (int dd = 0; dd < 24; ++dd) {
        V16 a, b0, b1;
        const int ad = dd * 32 + hi * 8;
        a.u[0] = *(const uint4*)(Qrow + ad);
        a.u[1] = *(const uint4*)(Qrow + ad + 16);
        const int db = dd * 32 + hi * 8;
        const __bf16* kp0 = Ks + (size_t)l15 * ND + db;
        const __bf16* kp1 = kp0 + 16 * ND;
        b0.u[0] = *(const uint4*)(kp0); b0.u[1] = *(const uint4*)(kp0 + 16);
        b1.u[0] = *(const uint4*)(kp1); b1.u[1] = *(const uint4*)(kp1 + 16);
        s0 = __builtin_amdgcn_wmma_f32_16x16x32_bf16(false, a.v, false, b0.v,
                                                     (short)0, s0, false, false);
        s1 = __builtin_amdgcn_wmma_f32_16x16x32_bf16(false, a.v, false, b1.v,
                                                     (short)0, s1, false, false);
      }
      // ---- scale + causal mask (lane holds col l15 of rows j,j+8)
      float sv[2][8];
      const int kg0 = kbase + l15, kg1 = kbase + 16 + l15;
      #pragma unroll
      for (int j = 0; j < 8; ++j) {
        const int qg = qb + j + hi8;
        sv[0][j] = (kg0 > qg) ? -1e30f : s0[j] * scale;
        sv[1][j] = (kg1 > qg) ? -1e30f : s1[j] * scale;
      }
      // ---- online softmax (row reductions within 16-lane halves)
      float alpha[8];
      #pragma unroll
      for (int j = 0; j < 8; ++j) {
        float mx = fmaxf(sv[0][j], sv[1][j]);
        mx = fmaxf(mx, __shfl_xor(mx, 1));
        mx = fmaxf(mx, __shfl_xor(mx, 2));
        mx = fmaxf(mx, __shfl_xor(mx, 4));
        mx = fmaxf(mx, __shfl_xor(mx, 8));
        const float mnew = fmaxf(m_run[j], mx);
        const float a = exp2f((m_run[j] - mnew) * LOG2E);
        const float p0 = exp2f((sv[0][j] - mnew) * LOG2E);
        const float p1 = exp2f((sv[1][j] - mnew) * LOG2E);
        sv[0][j] = p0; sv[1][j] = p1;
        float ps = p0 + p1;
        ps += __shfl_xor(ps, 1);
        ps += __shfl_xor(ps, 2);
        ps += __shfl_xor(ps, 4);
        ps += __shfl_xor(ps, 8);
        l_run[j] = l_run[j] * a + ps;
        m_run[j] = mnew;
        alpha[j] = a;
      }
      // ---- stage P (bf16) in wave-private LDS, then rebuild as A fragment
      __bf16* Pw = Ps + wv * 512;
      #pragma unroll
      for (int j = 0; j < 8; ++j) {
        Pw[(j + hi8) * 32 + l15]      = (__bf16)sv[0][j];
        Pw[(j + hi8) * 32 + 16 + l15] = (__bf16)sv[1][j];
      }
      asm volatile("s_wait_dscnt 0" ::: "memory");  // intra-wave LDS RAW
      V16 pa;
      pa.u[0] = *(const uint4*)(Pw + l15 * 32 + hi * 8);
      pa.u[1] = *(const uint4*)(Pw + l15 * 32 + hi * 8 + 16);
      // ---- rescale running O, then O += P * V  (48 WMMAs, K=32)
      #pragma unroll
      for (int ot = 0; ot < 48; ++ot) {
        #pragma unroll
        for (int j = 0; j < 8; ++j) oacc[ot][j] *= alpha[j];
        V16 vb;
        const __bf16* vp = Vs + (size_t)(ot * 16 + l15) * 32 + hi * 8;
        vb.u[0] = *(const uint4*)(vp);
        vb.u[1] = *(const uint4*)(vp + 16);
        oacc[ot] = __builtin_amdgcn_wmma_f32_16x16x32_bf16(false, pa.v, false, vb.v,
                                                           (short)0, oacc[ot], false, false);
      }
    }
    // next chunk's async loads must be in LDS, and all waves done reading `buf`
    asm volatile("s_wait_asynccnt 0" ::: "memory");
    __syncthreads();
  }

  // ---- epilogue: O / l, fp32 out
  float invl[8];
  #pragma unroll
  for (int j = 0; j < 8; ++j) invl[j] = 1.0f / l_run[j];
  float* orow = out + bofs;
  #pragma unroll
  for (int ot = 0; ot < 48; ++ot) {
    #pragma unroll
    for (int j = 0; j < 8; ++j)
      orow[(size_t)(qb + j + hi8) * ND + ot * 16 + l15] = oacc[ot][j] * invl[j];
  }
}

// ---------------------------------------------------------------- host
extern "C" void kernel_launch(void* const* d_in, const int* in_sizes, int n_in,
                              void* d_out, int out_size, void* d_ws, size_t ws_size,
                              hipStream_t stream) {
  const float* X  = (const float*)d_in[0];
  const float* Wq = (const float*)d_in[1];
  const float* Wk = (const float*)d_in[2];
  const float* Wv = (const float*)d_in[3];
  float* out = (float*)d_out;

  char* ws = (char*)d_ws;
  const size_t szX = (size_t)NB * NSEQ * ND * 2;    // 24MB
  const size_t szW = (size_t)ND * ND * 2;           // 1.125MB
  __bf16* Xb  = (__bf16*)(ws);
  __bf16* Qb  = (__bf16*)(ws + szX);
  __bf16* Kb  = (__bf16*)(ws + 2 * szX);
  __bf16* Vtb = (__bf16*)(ws + 3 * szX);
  __bf16* Wqb = (__bf16*)(ws + 4 * szX);
  __bf16* Wkb = (__bf16*)(ws + 4 * szX + szW);
  __bf16* Wvb = (__bf16*)(ws + 4 * szX + 2 * szW);

  // 1) fp32 -> bf16
  cvt_f32_bf16<<<(NB * NSEQ * ND / 8 + 255) / 256, 256, 0, stream>>>(X, Xb, NB * NSEQ * ND / 8);
  cvt_f32_bf16<<<(ND * ND / 8 + 255) / 256, 256, 0, stream>>>(Wq, Wqb, ND * ND / 8);
  cvt_f32_bf16<<<(ND * ND / 8 + 255) / 256, 256, 0, stream>>>(Wk, Wkb, ND * ND / 8);
  cvt_f32_bf16<<<(ND * ND / 8 + 255) / 256, 256, 0, stream>>>(Wv, Wvb, ND * ND / 8);

  // 2) Q,K,V projections: 3 * 1024 * 48 wave-tiles, 8 waves/block
  qkv_proj<<<18432, 256, 0, stream>>>(Xb, Wqb, Wkb, Wvb, Qb, Kb, Vtb);

  // 3) flash attention: 256 WGs (4 batches * 64 q-blocks), 200704B dynamic LDS
  flash_attn<<<NB * (NSEQ / 64), 128, SMEM_SZ, stream>>>(Qb, Kb, Vtb, out);
}
